// FrameHOIGraph_11527692222688
// MI455X (gfx1250) — compile-verified, matrix operand order checked by reading, not run on hardware
//
#include <hip/hip_runtime.h>

#define D      512
#define NTOK   16
#define BT     4096
#define FF     2048
#define MT2    32               // rows per block in FFN kernel
#define Y_ELEMS ((size_t)BT * NTOK * D)

typedef float  v8f   __attribute__((ext_vector_type(8)));
typedef float  v2f   __attribute__((ext_vector_type(2)));
typedef __bf16 v16bf __attribute__((ext_vector_type(16)));
typedef __bf16 v8bf  __attribute__((ext_vector_type(8)));

static __device__ inline v8f wmma_f32_k4(v2f a, v2f b, v8f c) {
  // D = A(16x4 f32) * B(4x16 f32) + C, exact fp32
  return __builtin_amdgcn_wmma_f32_16x16x4_f32(false, a, false, b, (short)0, c,
                                               false, false);
}
static __device__ inline v8f wmma_bf(v16bf a, v16bf b, v8f c) {
  // D = A(16x32 bf16) * B(32x16 bf16) + C(f32)
  return __builtin_amdgcn_wmma_f32_16x16x32_bf16(false, a, false, b, (short)0, c,
                                                 false, false);
}
// 16 contiguous bf16 -> fragment (two 16B vector loads)
static __device__ inline v16bf ld_frag16(const __bf16* p) {
  v8bf lo = *(const v8bf*)p;
  v8bf hi = *(const v8bf*)(p + 8);
  return __builtin_shufflevector(lo, hi, 0, 1, 2, 3, 4, 5, 6, 7, 8, 9, 10, 11,
                                 12, 13, 14, 15);
}
// two 8-element runs 16 elements apart (A-fragment pattern) -> fragment
static __device__ inline v16bf ld_frag_gap(const __bf16* p) {
  v8bf lo = *(const v8bf*)p;
  v8bf hi = *(const v8bf*)(p + 16);
  return __builtin_shufflevector(lo, hi, 0, 1, 2, 3, 4, 5, 6, 7, 8, 9, 10, 11,
                                 12, 13, 14, 15);
}
// K-offset of fragment element e within a 32-deep step (16-bit WMMA layout)
static __device__ inline int koff(int e, int half) {
  return (e & 7) + ((e >> 3) << 4) + half * 8;
}

// ---------------------------------------------------------------------------
// Kernel 1: per-batch attention + residual + LN1 (exact fp32 WMMA path).
// Writes h (post-LN1) into outy (scratch; kernel 2 overwrites with final y),
// and attn probabilities into outattn.
// ---------------------------------------------------------------------------
__global__ __launch_bounds__(128) void attn_ln1_kernel(
    const float* __restrict__ x, const unsigned char* __restrict__ mask,
    const float* __restrict__ g1, const float* __restrict__ b1,
    float* __restrict__ outy, float* __restrict__ outattn) {
  __shared__ __align__(16) float xs[NTOK * D];      // 32 KB
  __shared__ __align__(16) float hs[NTOK * D];      // 32 KB
  __shared__ float sc[4][NTOK * NTOK];              // 4 KB (per-wave partials)
  __shared__ float attn_s[NTOK * NTOK];             // 1 KB
  __shared__ float redS[NTOK][8];
  __shared__ float redQ[NTOK][8];

  const int b    = blockIdx.x;
  const int tid  = threadIdx.x;
  const int lane = tid & 31;
  const int wave = tid >> 5;
  const int nlo  = lane & 15;
  const int half = lane >> 4;

  const float* xb = x + (size_t)b * (NTOK * D);

  // ---- stage x tile (16x512) into LDS, coalesced float4 ----
  for (int i = tid; i < NTOK * D / 4; i += 128)
    ((float4*)xs)[i] = ((const float4*)xb)[i];
  __syncthreads();

  // ---- scores = X * X^T / sqrt(D), K split across 4 waves ----
  {
    v8f acc = {};
    const int kw = wave * 128;
#pragma unroll 4
    for (int kk = 0; kk < 32; ++kk) {
      const int k0 = kw + kk * 4 + half * 2;
      v2f a;
      a.x = xs[nlo * D + k0];
      a.y = xs[nlo * D + k0 + 1];
      // X.X^T symmetry: A fragment (M=lane) == B fragment (N=lane)
      acc = wmma_f32_k4(a, a, acc);
    }
#pragma unroll
    for (int r = 0; r < 8; ++r)
      sc[wave][(r + half * 8) * NTOK + nlo] = acc[r];
  }
  __syncthreads();

  // ---- combine partials, mask, softmax (one thread per query row) ----
  if (tid < NTOK) {
    const unsigned char* mrow = mask + (size_t)b * NTOK * NTOK + tid * NTOK;
    float srow[NTOK];
    float mx = -__builtin_inff();
#pragma unroll
    for (int n = 0; n < NTOK; ++n) {
      const int idx = tid * NTOK + n;
      float s = (sc[0][idx] + sc[1][idx] + sc[2][idx] + sc[3][idx]) *
                0.0441941738241592f;  // 1/sqrt(512)
      s = mrow[n] ? -__builtin_inff() : s;
      srow[n] = s;
      mx = fmaxf(mx, s);
    }
    float sum = 0.f;
#pragma unroll
    for (int n = 0; n < NTOK; ++n) {
      const float e = __expf(srow[n] - mx);
      srow[n] = e;
      sum += e;
    }
    const float inv = 1.f / sum;
    float* oa = outattn + (size_t)b * NTOK * NTOK + tid * NTOK;
#pragma unroll
    for (int n = 0; n < NTOK; ++n) {
      const float p = srow[n] * inv;
      attn_s[tid * NTOK + n] = p;
      oa[n] = p;
    }
  }
  __syncthreads();

  // ---- out = attn @ X ; h_pre = x + out (each wave owns 8 D-chunks) ----
  for (int ci = 0; ci < 8; ++ci) {
    const int c = wave * 8 + ci;
    v8f acc = {};
#pragma unroll
    for (int ks = 0; ks < 4; ++ks) {
      const int kb = ks * 4 + half * 2;
      v2f a, bb;
      a.x  = attn_s[nlo * NTOK + kb];
      a.y  = attn_s[nlo * NTOK + kb + 1];
      bb.x = xs[kb * D + c * 16 + nlo];
      bb.y = xs[(kb + 1) * D + c * 16 + nlo];
      acc = wmma_f32_k4(a, bb, acc);
    }
#pragma unroll
    for (int r = 0; r < 8; ++r) {
      const int m = r + half * 8;
      const int d = c * 16 + nlo;
      hs[m * D + d] = xs[m * D + d] + acc[r];
    }
  }
  __syncthreads();

  // ---- LN1: 16 rows, 8 threads per row ----
  {
    const int m = tid >> 3, s8 = tid & 7;
    float sum = 0.f, sq = 0.f;
    for (int j = 0; j < 64; ++j) {
      const float v = hs[m * D + s8 * 64 + j];
      sum += v;
      sq += v * v;
    }
    redS[m][s8] = sum;
    redQ[m][s8] = sq;
    __syncthreads();
    float ts = 0.f, tq = 0.f;
#pragma unroll
    for (int j = 0; j < 8; ++j) { ts += redS[m][j]; tq += redQ[m][j]; }
    const float mu   = ts * (1.f / D);
    const float rstd = rsqrtf(tq * (1.f / D) - mu * mu + 1e-5f);
    float* hdst = outy + (size_t)b * (NTOK * D) + m * D;
    for (int j = 0; j < 64; ++j) {
      const int d = s8 * 64 + j;
      hdst[d] = (hs[m * D + d] - mu) * rstd * g1[d] + b1[d];
    }
  }
}

// ---------------------------------------------------------------------------
// Weight pre-pack: f32 W1/W2 -> bf16 in B-fragment-major order.
// W1s block (nt,ks): 32 lanes x 16 bf16, lane-contiguous (1 KB per block).
//   W1s[((nt*16+ks)*512) + lane*16 + e] = W1[(ks*32 + koff(e))*2048 + nt*16+nlo]
// W2s block (nt2,kst): same shape, 32 nt2-tiles x 64 k-steps.
// ---------------------------------------------------------------------------
__global__ __launch_bounds__(256) void pack_weights_kernel(
    const float* __restrict__ W1, const float* __restrict__ W2,
    __bf16* __restrict__ W1s, __bf16* __restrict__ W2s) {
  const int idx  = blockIdx.x * 256 + threadIdx.x;  // 0 .. 2^20-1
  const int e    = idx & 15;
  const int lane = (idx >> 4) & 31;
  const int blk  = idx >> 9;
  const int half = lane >> 4;
  const int nlo  = lane & 15;
  const int ko   = koff(e, half);
  {  // W1s: blk = nt*16 + ks   (nt: 128, ks: 16)
    const int ks = blk & 15;
    const int nt = blk >> 4;
    W1s[idx] = (__bf16)W1[(size_t)(ks * 32 + ko) * FF + nt * 16 + nlo];
  }
  {  // W2s: blk = nt2*64 + kst (nt2: 32, kst: 64)
    const int kst = blk & 63;
    const int nt2 = blk >> 6;
    W2s[idx] = (__bf16)W2[(size_t)(kst * 32 + ko) * D + nt2 * 16 + nlo];
  }
}

// ---------------------------------------------------------------------------
// Kernel 2 (fast path): FFN with pre-swizzled bf16 weights.
// B fragment = two global_load_b128; A fragment = two ds_load_b128.
// ---------------------------------------------------------------------------
__global__ __launch_bounds__(256) void ffn_ln2_swz_kernel(
    const __bf16* __restrict__ W1s, const float* __restrict__ bW1,
    const __bf16* __restrict__ W2s, const float* __restrict__ bW2,
    const float* __restrict__ g2, const float* __restrict__ b2,
    float* __restrict__ outy) {
  __shared__ __align__(16) __bf16 hb[MT2 * D];      // 32 KB  h tile (bf16)
  __shared__ __align__(16) __bf16 t1b[MT2 * 128];   // 8 KB   relu(h@W1) chunk
  __shared__ __align__(16) float vbuf[MT2 * D];     // 64 KB  h + ff for LN
  __shared__ float redS[MT2][8];
  __shared__ float redQ[MT2][8];

  const int tid  = threadIdx.x;
  const int lane = tid & 31;
  const int wave = tid >> 5;
  const int nlo  = lane & 15;
  const int half = lane >> 4;

  const size_t row0 = (size_t)blockIdx.x * MT2;
  const float* hsrc = outy + row0 * D;

  // ---- stage h tile, convert to bf16 ----
  for (int i = tid; i < MT2 * D / 4; i += 256) {
    const float4 v = ((const float4*)hsrc)[i];
    hb[4 * i + 0] = (__bf16)v.x;
    hb[4 * i + 1] = (__bf16)v.y;
    hb[4 * i + 2] = (__bf16)v.z;
    hb[4 * i + 3] = (__bf16)v.w;
  }

  // ---- persistent f32 accumulators for ff = t1 @ W2 (+bW2 folded in) ----
  v8f acc2[4][2];
#pragma unroll
  for (int q = 0; q < 4; ++q) {
    const float bv = bW2[(wave * 4 + q) * 16 + nlo];
#pragma unroll
    for (int mi = 0; mi < 2; ++mi)
#pragma unroll
      for (int r = 0; r < 8; ++r) acc2[q][mi][r] = bv;
  }
  __syncthreads();

  for (int kc = 0; kc < 16; ++kc) {   // 16 chunks of 128 hidden columns
    // ---- phase 1: t1 = relu(h @ W1[:, chunk] + bW1), wave owns 16 cols ----
    const int nt = kc * 8 + wave;           // n-tile in 0..127
    const int c0 = nt * 16;
    v8f p0, p1;
    {
      const float bv1 = bW1[c0 + nlo];
#pragma unroll
      for (int r = 0; r < 8; ++r) { p0[r] = bv1; p1[r] = bv1; }
    }
    const __bf16* w1base = W1s + ((size_t)nt << 13) + lane * 16;  // nt*16*512
#pragma unroll 2
    for (int ks = 0; ks < 16; ++ks) {       // K = 512 in steps of 32
      const int kb = ks * 32;
      const v16bf bfr = ld_frag16(w1base + ((size_t)ks << 9));
      const v16bf a0  = ld_frag_gap(&hb[nlo * D + kb + half * 8]);
      const v16bf a1  = ld_frag_gap(&hb[(16 + nlo) * D + kb + half * 8]);
      p0 = wmma_bf(a0, bfr, p0);
      p1 = wmma_bf(a1, bfr, p1);
    }
#pragma unroll
    for (int r = 0; r < 8; ++r) {
      const int cc = wave * 16 + nlo;
      t1b[(r + half * 8) * 128 + cc]      = (__bf16)fmaxf(p0[r], 0.f);
      t1b[(16 + r + half * 8) * 128 + cc] = (__bf16)fmaxf(p1[r], 0.f);
    }
    __syncthreads();

    // ---- phase 2: acc2 += t1 @ W2[chunk, :]  (A hoisted across n-tiles) ----
#pragma unroll
    for (int ks2 = 0; ks2 < 4; ++ks2) {     // K = 128 in steps of 32
      const int kb  = ks2 * 32;
      const int kst = kc * 4 + ks2;         // global k-step 0..63
      const v16bf a20 = ld_frag_gap(&t1b[nlo * 128 + kb + half * 8]);
      const v16bf a21 = ld_frag_gap(&t1b[(16 + nlo) * 128 + kb + half * 8]);
#pragma unroll
      for (int q = 0; q < 4; ++q) {
        const int nt2 = wave * 4 + q;       // n-tile in 0..31
        const v16bf bfr2 =
            ld_frag16(W2s + (((size_t)nt2 * 64 + kst) << 9) + lane * 16);
        acc2[q][0] = wmma_bf(a20, bfr2, acc2[q][0]);
        acc2[q][1] = wmma_bf(a21, bfr2, acc2[q][1]);
      }
    }
    __syncthreads();
  }

  // ---- epilogue: v = h + ff(+bW2), LN2, write y ----
#pragma unroll
  for (int q = 0; q < 4; ++q)
#pragma unroll
    for (int mi = 0; mi < 2; ++mi)
#pragma unroll
      for (int r = 0; r < 8; ++r) {
        const int m = mi * 16 + r + half * 8;
        const int d = (wave * 4 + q) * 16 + nlo;
        vbuf[m * D + d] = acc2[q][mi][r];
      }
  __syncthreads();
  for (int i = tid; i < MT2 * D; i += 256) vbuf[i] += hsrc[i];
  __syncthreads();

  const int m = tid >> 3, s8 = tid & 7;
  float sum = 0.f, sq = 0.f;
  for (int j = 0; j < 64; ++j) {
    const float v = vbuf[m * D + s8 * 64 + j];
    sum += v;
    sq += v * v;
  }
  redS[m][s8] = sum;
  redQ[m][s8] = sq;
  __syncthreads();
  float ts = 0.f, tq = 0.f;
#pragma unroll
  for (int j = 0; j < 8; ++j) { ts += redS[m][j]; tq += redQ[m][j]; }
  const float mu   = ts * (1.f / D);
  const float rstd = rsqrtf(tq * (1.f / D) - mu * mu + 1e-5f);
  float* ydst = outy + row0 * D + m * D;
  for (int j = 0; j < 64; ++j) {
    const int d = s8 * 64 + j;
    ydst[d] = (vbuf[m * D + d] - mu) * rstd * g2[d] + b2[d];
  }
}

// ---------------------------------------------------------------------------
// Kernel 2 (fallback, no workspace): converts W fragments from f32 on the fly.
// ---------------------------------------------------------------------------
__global__ __launch_bounds__(256) void ffn_ln2_direct_kernel(
    const float* __restrict__ W1, const float* __restrict__ bW1,
    const float* __restrict__ W2, const float* __restrict__ bW2,
    const float* __restrict__ g2, const float* __restrict__ b2,
    float* __restrict__ outy) {
  __shared__ __align__(16) __bf16 hb[MT2 * D];
  __shared__ __align__(16) __bf16 t1b[MT2 * 128];
  __shared__ __align__(16) float vbuf[MT2 * D];
  __shared__ float redS[MT2][8];
  __shared__ float redQ[MT2][8];

  const int tid  = threadIdx.x;
  const int lane = tid & 31;
  const int wave = tid >> 5;
  const int nlo  = lane & 15;
  const int half = lane >> 4;

  const size_t row0 = (size_t)blockIdx.x * MT2;
  const float* hsrc = outy + row0 * D;

  for (int i = tid; i < MT2 * D / 4; i += 256) {
    const float4 v = ((const float4*)hsrc)[i];
    hb[4 * i + 0] = (__bf16)v.x;
    hb[4 * i + 1] = (__bf16)v.y;
    hb[4 * i + 2] = (__bf16)v.z;
    hb[4 * i + 3] = (__bf16)v.w;
  }

  v8f acc2[4][2];
#pragma unroll
  for (int q = 0; q < 4; ++q) {
    const float bv = bW2[(wave * 4 + q) * 16 + nlo];
#pragma unroll
    for (int mi = 0; mi < 2; ++mi)
#pragma unroll
      for (int r = 0; r < 8; ++r) acc2[q][mi][r] = bv;
  }
  __syncthreads();

  for (int kc = 0; kc < 16; ++kc) {
    const int c0 = kc * 128 + wave * 16;
    v8f p0, p1;
    {
      const float bv1 = bW1[c0 + nlo];
#pragma unroll
      for (int r = 0; r < 8; ++r) { p0[r] = bv1; p1[r] = bv1; }
    }
    for (int ks = 0; ks < 16; ++ks) {
      const int kb = ks * 32;
      v16bf bfr;
#pragma unroll
      for (int e = 0; e < 16; ++e)
        bfr[e] = (__bf16)W1[(size_t)(kb + koff(e, half)) * FF + c0 + nlo];
      const v16bf a0 = ld_frag_gap(&hb[nlo * D + kb + half * 8]);
      const v16bf a1 = ld_frag_gap(&hb[(16 + nlo) * D + kb + half * 8]);
      p0 = wmma_bf(a0, bfr, p0);
      p1 = wmma_bf(a1, bfr, p1);
    }
#pragma unroll
    for (int r = 0; r < 8; ++r) {
      const int cc = wave * 16 + nlo;
      t1b[(r + half * 8) * 128 + cc]      = (__bf16)fmaxf(p0[r], 0.f);
      t1b[(16 + r + half * 8) * 128 + cc] = (__bf16)fmaxf(p1[r], 0.f);
    }
    __syncthreads();

#pragma unroll
    for (int ks2 = 0; ks2 < 4; ++ks2) {
      const int kb = ks2 * 32;
      const v16bf a20 = ld_frag_gap(&t1b[nlo * 128 + kb + half * 8]);
      const v16bf a21 = ld_frag_gap(&t1b[(16 + nlo) * 128 + kb + half * 8]);
#pragma unroll
      for (int q = 0; q < 4; ++q) {
        const int n0 = (wave * 4 + q) * 16;
        v16bf bfr2;
#pragma unroll
        for (int e = 0; e < 16; ++e)
          bfr2[e] =
              (__bf16)W2[(size_t)(kc * 128 + kb + koff(e, half)) * D + n0 + nlo];
        acc2[q][0] = wmma_bf(a20, bfr2, acc2[q][0]);
        acc2[q][1] = wmma_bf(a21, bfr2, acc2[q][1]);
      }
    }
    __syncthreads();
  }

#pragma unroll
  for (int q = 0; q < 4; ++q)
#pragma unroll
    for (int mi = 0; mi < 2; ++mi)
#pragma unroll
      for (int r = 0; r < 8; ++r) {
        const int m = mi * 16 + r + half * 8;
        const int d = (wave * 4 + q) * 16 + nlo;
        vbuf[m * D + d] = acc2[q][mi][r];
      }
  __syncthreads();
  for (int i = tid; i < MT2 * D; i += 256) vbuf[i] += hsrc[i];
  __syncthreads();

  const int m = tid >> 3, s8 = tid & 7;
  float sum = 0.f, sq = 0.f;
  for (int j = 0; j < 64; ++j) {
    const float v = vbuf[m * D + s8 * 64 + j];
    sum += v;
    sq += v * v;
  }
  redS[m][s8] = sum;
  redQ[m][s8] = sq;
  __syncthreads();
  float ts = 0.f, tq = 0.f;
#pragma unroll
  for (int j = 0; j < 8; ++j) { ts += redS[m][j]; tq += redQ[m][j]; }
  const float mu   = ts * (1.f / D);
  const float rstd = rsqrtf(tq * (1.f / D) - mu * mu + 1e-5f);
  float* ydst = outy + row0 * D + m * D;
  for (int j = 0; j < 64; ++j) {
    const int d = s8 * 64 + j;
    ydst[d] = (vbuf[m * D + d] - mu) * rstd * g2[d] + b2[d];
  }
}

// ---------------------------------------------------------------------------
extern "C" void kernel_launch(void* const* d_in, const int* in_sizes, int n_in,
                              void* d_out, int out_size, void* d_ws,
                              size_t ws_size, hipStream_t stream) {
  (void)in_sizes; (void)n_in; (void)out_size;
  const float*         x    = (const float*)d_in[0];
  const unsigned char* mask = (const unsigned char*)d_in[1];  // bool tensor
  const float*         g1   = (const float*)d_in[2];
  const float*         b1   = (const float*)d_in[3];
  const float*         g2   = (const float*)d_in[4];
  const float*         b2   = (const float*)d_in[5];
  const float*         W1   = (const float*)d_in[6];
  const float*         bW1  = (const float*)d_in[7];
  const float*         W2   = (const float*)d_in[8];
  const float*         bW2  = (const float*)d_in[9];

  float* outy    = (float*)d_out;            // y: [BT, 16, 512]
  float* outattn = outy + Y_ELEMS;           // attn: [BT, 16, 16]

  attn_ln1_kernel<<<BT, 128, 0, stream>>>(x, mask, g1, b1, outy, outattn);

  const size_t nW = (size_t)D * FF;          // 1M elements each
  if (ws_size >= 2 * nW * sizeof(__bf16)) {
    __bf16* W1s = (__bf16*)d_ws;
    __bf16* W2s = W1s + nW;
    pack_weights_kernel<<<(int)(nW / 256), 256, 0, stream>>>(W1, W2, W1s, W2s);
    ffn_ln2_swz_kernel<<<(BT * NTOK) / MT2, 256, 0, stream>>>(W1s, bW1, W2s,
                                                              bW2, g2, b2,
                                                              outy);
  } else {
    ffn_ln2_direct_kernel<<<(BT * NTOK) / MT2, 256, 0, stream>>>(W1, bW1, W2,
                                                                 bW2, g2, b2,
                                                                 outy);
  }
}
